// PatchEncoder_4526895530286
// MI455X (gfx1250) — matrix operand order repaired and checked by reading, not compiled, not run on hardware
//
#include <hip/hip_runtime.h>

// ---- CDNA5 WMMA fragment types (wave32) ----
typedef __attribute__((ext_vector_type(16))) __bf16 v16bf;
typedef __attribute__((ext_vector_type(8)))  float  v8f;

#define IMG     256
#define PATCH   16
#define CH      3
#define DIM     768                 // K == D == 768
#define MROWS   16384               // B * num_patches
#define ROWF    (IMG * CH)          // 768 floats per image pixel row
#define SEGF    (PATCH * CH)        // 48 floats per patch pixel-row segment

#define BM  128
#define BN  64
#define BKS 96                      // staged K per iteration = 2 pixel rows
#define NPR 8                       // 768 / 96 stage iterations

// pack two f32 -> two bf16 (round-half-up) in one dword via v_perm_b32
__device__ __forceinline__ unsigned int pk_bf16(float lo, float hi) {
    unsigned int u0 = __builtin_bit_cast(unsigned int, lo) + 0x8000u;
    unsigned int u1 = __builtin_bit_cast(unsigned int, hi) + 0x8000u;
    return __builtin_amdgcn_perm(u1, u0, 0x07060302u);  // {u1[31:16], u0[31:16]}
}

__global__ __launch_bounds__(256) void patch_encoder_wmma(
    const float* __restrict__ x,     // [64,256,256,3]
    const float* __restrict__ W,     // [768,768]
    const float* __restrict__ bias,  // [768]
    const float* __restrict__ pos,   // [256,768]
    float* __restrict__ out)         // [16384,768]
{
    __shared__ __align__(16) unsigned short sA[BM * BKS];   // [row][k]  bf16
    __shared__ __align__(16) unsigned short sBt[BN * BKS];  // [col][k]  bf16 (W^T)

    const int tid  = threadIdx.x;
    const int lane = tid & 31;
    const int wave = tid >> 5;
    const int wm   = wave & 3;   // 4 waves over M
    const int wn   = wave >> 2;  // 2 waves over N

    const int tileM = blockIdx.y * BM;
    const int tileN = blockIdx.x * BN;

    // ---- A staging: one thread = one 48-float contiguous pixel-row segment ----
    const int a_row = tid >> 1;          // 0..127 (patch row within tile)
    const int a_seg = tid & 1;           // which of the 2 pixel rows this stage
    const int m_g   = tileM + a_row;
    const int bi    = m_g >> 8;
    const int pidx  = m_g & 255;
    const int hp    = pidx >> 4;
    const int wp    = pidx & 15;
    // &x[bi, hp*16, wp*16, 0] : start of this patch's py=0 segment
    const float* abase = x + ((bi * IMG + hp * PATCH) * IMG + wp * PATCH) * CH;

    // ---- B staging: k-pair granularity so LDS stores are packed b32 ----
    const int b_cg = (tid & 15) * 4;     // column group base: 0,4,..,60
    const int b_kp = (tid >> 4) * 2;     // even k-row base:   0,2,..,30

    v8f acc[2][2];
    #pragma unroll
    for (int i = 0; i < 2; ++i)
        #pragma unroll
        for (int j = 0; j < 2; ++j)
            acc[i][j] = {};

    const int kgrp = lane >> 4;  // lane half (0/1)
    const int l15  = lane & 15;

    for (int pr = 0; pr < NPR; ++pr) {        // k0 = pr * 96
        // ============ stage A: 48 contiguous floats -> 48 bf16 in LDS ============
        {
            const float* src = abase + (2 * pr + a_seg) * ROWF;   // py = 2*pr + seg
            float4 f[12];
            #pragma unroll
            for (int i = 0; i < 12; ++i)
                f[i] = *(const float4*)(src + 4 * i);
            if (pr + 1 < NPR)   // prefetch next stage's x segment (dominant stream)
                __builtin_prefetch((const void*)(src + 2 * ROWF), 0, 0);
            uint2* dst = (uint2*)&sA[a_row * BKS + a_seg * SEGF];
            #pragma unroll
            for (int i = 0; i < 12; ++i) {
                uint2 p;
                p.x = pk_bf16(f[i].x, f[i].y);
                p.y = pk_bf16(f[i].z, f[i].w);
                dst[i] = p;
            }
        }
        // ===== stage B: 96x64 W slice -> bf16 transposed, packed b32 stores =====
        {
            #pragma unroll
            for (int r = 0; r < 3; ++r) {
                const int krow = b_kp + 32 * r;                  // even, 0..94
                const float* w0p = W + (pr * BKS + krow) * DIM + tileN + b_cg;
                float4 w0 = *(const float4*)(w0p);
                float4 w1 = *(const float4*)(w0p + DIM);
                // column c gets k-pair (krow, krow+1) packed into one dword
                *(unsigned int*)&sBt[(b_cg + 0) * BKS + krow] = pk_bf16(w0.x, w1.x);
                *(unsigned int*)&sBt[(b_cg + 1) * BKS + krow] = pk_bf16(w0.y, w1.y);
                *(unsigned int*)&sBt[(b_cg + 2) * BKS + krow] = pk_bf16(w0.z, w1.z);
                *(unsigned int*)&sBt[(b_cg + 3) * BKS + krow] = pk_bf16(w0.w, w1.w);
            }
            if (pr + 1 < NPR)   // prefetch next stage's W rows
                __builtin_prefetch(
                    (const void*)(W + ((pr + 1) * BKS + b_kp) * DIM + tileN + b_cg),
                    0, 0);
        }
        __syncthreads();

        // ============ 3 WMMA k-steps over the staged 96 ============
        #pragma unroll
        for (int ks = 0; ks < 3; ++ks) {
            const int kb = ks * 32;
            // A 16x32 bf16 frag: lane l = row (l&15); chunks [8*kgrp,+8), [16+8*kgrp,+8)
            v16bf afrag[2];
            #pragma unroll
            for (int i = 0; i < 2; ++i) {
                const unsigned short* ar = &sA[(wm * 32 + i * 16 + l15) * BKS + kb];
                union { v16bf v; uint4 u[2]; } t;
                t.u[0] = *(const uint4*)(ar + 8 * kgrp);
                t.u[1] = *(const uint4*)(ar + 16 + 8 * kgrp);
                afrag[i] = t.v;
            }
            // B 32x16 bf16 frag: lane l = col (l&15); chunks [8*kgrp,+8), [16+8*kgrp,+8)
            v16bf bfrag[2];
            #pragma unroll
            for (int j = 0; j < 2; ++j) {
                const unsigned short* br =
                    &sBt[(wn * 32 + j * 16 + l15) * BKS + kb + 8 * kgrp];
                union { v16bf v; uint4 u[2]; } t;
                t.u[0] = *(const uint4*)(br);
                t.u[1] = *(const uint4*)(br + 16);
                bfrag[j] = t.v;
            }
            #pragma unroll
            for (int i = 0; i < 2; ++i)
                #pragma unroll
                for (int j = 0; j < 2; ++j)
                    acc[i][j] = __builtin_amdgcn_wmma_f32_16x16x32_bf16(
                        false, afrag[i], false, bfrag[j],
                        (short)0, acc[i][j], false, false);
        }
        __syncthreads();
    }

    // ============ epilogue: + bias + pos_emb, coalesced stores ============
    #pragma unroll
    for (int i = 0; i < 2; ++i) {
        #pragma unroll
        for (int j = 0; j < 2; ++j) {
            const int ncol = tileN + wn * 32 + j * 16 + l15;
            const float bn = bias[ncol];
            #pragma unroll
            for (int v = 0; v < 8; ++v) {
                const int mrow = tileM + wm * 32 + i * 16 + v + 8 * kgrp;
                out[mrow * DIM + ncol] =
                    acc[i][j][v] + bn + pos[(mrow & 255) * DIM + ncol];
            }
        }
    }
}

extern "C" void kernel_launch(void* const* d_in, const int* in_sizes, int n_in,
                              void* d_out, int out_size, void* d_ws, size_t ws_size,
                              hipStream_t stream) {
    const float* x    = (const float*)d_in[0];
    const float* W    = (const float*)d_in[1];
    const float* bias = (const float*)d_in[2];
    const float* pos  = (const float*)d_in[3];
    float* out        = (float*)d_out;

    dim3 grid(DIM / BN, MROWS / BM);  // (12, 128)
    dim3 block(256);
    patch_encoder_wmma<<<grid, block, 0, stream>>>(x, W, bias, pos, out);
}
